// VoxelBackBoneDeconv_6451040878700
// MI455X (gfx1250) — matrix operand order repaired and checked by reading, not compiled, not run on hardware
//
#include <hip/hip_runtime.h>

// ---------------------------------------------------------------------------
// CDNA5 (gfx1250) wave32 WMMA implicit-GEMM 3D conv backbone.
// Tensors are kept in bf16 channels-last layout [z][y][x][c] between layers.
// Each tensor pool has a zeroed 256B prefix: out-of-bounds / invalid taps load
// from negative offsets (branchless cndmask select, no exec-mask diamonds).
// K-loop is explicitly double-buffered: chunk q+1 loads issue before chunk q's
// WMMA so VMEM latency overlaps the matrix pipe.
// ---------------------------------------------------------------------------

typedef __attribute__((ext_vector_type(16))) __bf16 v16bf;
typedef __attribute__((ext_vector_type(8)))  float  v8f;

#define ZPAD_EL 128   // zero-prefix size in bf16 elements (256 bytes)

union Frag {
  v16bf v;
  uint4 u4[2];
  unsigned int u[8];
};

__device__ __forceinline__ unsigned short f2bf(float f) {
  unsigned int b = __float_as_uint(f);
  b += 0x7fffu + ((b >> 16) & 1u);        // round-to-nearest-even
  return (unsigned short)(b >> 16);
}

// ---------------------------------------------------------------------------
// Scatter sparse voxel features into dense bf16 channels-last grid + mask m0.
// ---------------------------------------------------------------------------
__global__ void scatter_kernel(const float* __restrict__ feats,
                               const int* __restrict__ coords,
                               unsigned short* __restrict__ dense,  // [41][256][256][4] bf16
                               unsigned char* __restrict__ m0,      // [41][256][256]
                               int n) {
  int i = blockIdx.x * blockDim.x + threadIdx.x;
  if (i >= n) return;
  int z = coords[4 * i + 1], y = coords[4 * i + 2], x = coords[4 * i + 3];
  size_t p = ((size_t)z * 256 + y) * 256 + x;
  m0[p] = 1;
#pragma unroll
  for (int c = 0; c < 4; ++c) dense[p * 4 + c] = f2bf(feats[4 * i + c]);
}

// ---------------------------------------------------------------------------
// Mask dilation: out = (conv(in, ones3x3x3, mode) > 0).
// MODE: 0 = stride1 pad1, 1 = stride2 pad1, 2 = transposed stride2 pad1.
// ---------------------------------------------------------------------------
template <int MODE>
__global__ void mask_dilate(const unsigned char* __restrict__ in,
                            unsigned char* __restrict__ out,
                            int Din, int Hin, int Win,
                            int Dout, int Hout, int Wout) {
  long idx = (long)blockIdx.x * blockDim.x + threadIdx.x;
  long total = (long)Dout * Hout * Wout;
  if (idx >= total) return;
  int x = (int)(idx % Wout);
  long t = idx / Wout;
  int y = (int)(t % Hout);
  int z = (int)(t / Hout);
  int any = 0;
#pragma unroll
  for (int dz = 0; dz < 3; ++dz) {
    int iz; bool okz;
    if (MODE == 2) { int u = z + dz - 1; iz = u >> 1; okz = ((u & 1) == 0) && ((unsigned)iz < (unsigned)Din); }
    else { const int S = (MODE == 1) ? 2 : 1; iz = z * S + dz - 1; okz = (unsigned)iz < (unsigned)Din; }
    if (!okz) continue;
#pragma unroll
    for (int dy = 0; dy < 3; ++dy) {
      int iy; bool oky;
      if (MODE == 2) { int u = y + dy - 1; iy = u >> 1; oky = ((u & 1) == 0) && ((unsigned)iy < (unsigned)Hin); }
      else { const int S = (MODE == 1) ? 2 : 1; iy = y * S + dy - 1; oky = (unsigned)iy < (unsigned)Hin; }
      if (!oky) continue;
#pragma unroll
      for (int dx = 0; dx < 3; ++dx) {
        int ix; bool okx;
        if (MODE == 2) { int u = x + dx - 1; ix = u >> 1; okx = ((u & 1) == 0) && ((unsigned)ix < (unsigned)Win); }
        else { const int S = (MODE == 1) ? 2 : 1; ix = x * S + dx - 1; okx = (unsigned)ix < (unsigned)Win; }
        if (okx && in[((long)iz * Hin + iy) * Win + ix]) any = 1;
      }
    }
  }
  out[idx] = (unsigned char)any;
}

// ---------------------------------------------------------------------------
// Repack conv weights (fp32, OIDHW or IODHW-flipped for tconv) into the exact
// per-lane bf16 B-fragment layout of v_wmma_f32_16x16x32_bf16:
//   frag[((nc*KCH + q)*32 + lane)*16 + i],  k = q*32 + (i&7) + (lane>=16)*8 + (i>=8)*16,
//   n = lane&15,  k = tap*C_IN + c.
// ---------------------------------------------------------------------------
__global__ void repack_weights(const float* __restrict__ w,
                               unsigned short* __restrict__ frag,
                               int C_IN, int C_OUT, int KCH, int NCH,
                               int transposed) {
  int idx = blockIdx.x * blockDim.x + threadIdx.x;
  int total = NCH * KCH * 512;
  if (idx >= total) return;
  int i    = idx & 15;
  int lane = (idx >> 4) & 31;
  int t2   = idx >> 9;
  int q    = t2 % KCH;
  int nc   = t2 / KCH;
  int kb   = (i & 7) + ((lane >> 4) << 3) + ((i >> 3) << 4);
  int k    = (q << 5) + kb;
  int tap  = k / C_IN;
  int c    = k % C_IN;
  int o    = (nc << 4) + (lane & 15);
  float val = 0.f;
  if (tap < 27) {
    if (transposed) val = w[((size_t)c * C_OUT + o) * 27 + (26 - tap)];  // swapaxes+flip baked in
    else            val = w[((size_t)o * C_IN + c) * 27 + tap];
  }
  frag[idx] = f2bf(val);
}

// ---------------------------------------------------------------------------
// Fused conv3d (implicit GEMM, WMMA bf16) + scale/shift + ReLU + mask.
// MODE: 0 = stride1 pad1, 1 = stride2 pad1, 2 = transposed-conv stride2 pad1.
// Block: 32*(C_OUT/16) threads; wave w computes output channels [16w,16w+16).
// Each block computes 16 consecutive x outputs at one (z,y).
// K ordering: k = tap*C_IN + c (channels contiguous in memory -> b128 loads).
// Invalid taps read the zeroed prefix at negative offsets (pure cndmask).
// ---------------------------------------------------------------------------
template <int C_IN, int C_OUT, int MODE, bool FINAL>
__global__ __launch_bounds__(2 * C_OUT)
void conv3d_wmma(const unsigned short* __restrict__ in,   // bf16 [Din][Hin][Win][C_IN], zeroed 256B prefix before it
                 const unsigned int* __restrict__ bfrag,  // packed B fragments
                 const float* __restrict__ scale,
                 const float* __restrict__ shift,
                 const unsigned char* __restrict__ mask,  // [Dout][Hout][Wout]
                 unsigned short* __restrict__ obf,        // bf16 channels-last out
                 float* __restrict__ of32,                // fp32 NCDHW out (FINAL)
                 int Din, int Hin, int Win,
                 int Dout, int Hout, int Wout) {
  constexpr int CLOG = (C_IN == 4) ? 2 : (C_IN == 16) ? 4 : (C_IN == 32) ? 5 : 6;
  constexpr int KCH  = (27 * C_IN + 31) / 32;
  const int lane = threadIdx.x & 31;
  const int wv   = threadIdx.x >> 5;   // n-chunk
  const int tilesX = (Wout + 15) >> 4;
  int b  = blockIdx.x;
  int tx = b % tilesX;
  int t  = b / tilesX;
  int y  = t % Hout;
  int z  = t / Hout;
  int x0 = tx << 4;
  const int hi = lane >> 4;
  const int ox = x0 + (lane & 15);     // output x for this lane's A rows

  v8f acc = {0.f, 0.f, 0.f, 0.f, 0.f, 0.f, 0.f, 0.f};

  const unsigned int* bp = bfrag + (((size_t)wv * KCH * 32 + lane) << 3);

  // Branchless per-tap base address: one 32-bit cndmask on the element offset.
  auto tapAddr = [&](int dz, int dy, int dx, int coff) -> const unsigned short* {
    int iz, iy, ix;
    bool ok;
    if (MODE == 2) {
      int uz = z + dz - 1, uy = y + dy - 1, ux = ox + dx - 1;
      iz = uz >> 1; iy = uy >> 1; ix = ux >> 1;
      ok = ((uz & 1) == 0) && ((uy & 1) == 0) && ((ux & 1) == 0) &&
           ((unsigned)iz < (unsigned)Din) && ((unsigned)iy < (unsigned)Hin) &&
           ((unsigned)ix < (unsigned)Win);
    } else {
      const int S = (MODE == 1) ? 2 : 1;
      iz = z * S + dz - 1; iy = y * S + dy - 1; ix = ox * S + dx - 1;
      ok = ((unsigned)iz < (unsigned)Din) && ((unsigned)iy < (unsigned)Hin) &&
           ((unsigned)ix < (unsigned)Win);
    }
    int el = (((iz * Hin + iy) * Win + ix) << CLOG) + coff;   // element index
    el = ok ? el : -ZPAD_EL;                                  // branchless: zero prefix
    return in + (long)el;
  };

  // Load A and B fragments for K-chunk q (q is a compile-time constant under
  // full unroll, so all tap math constant-folds).
  auto loadAB = [&](int q, Frag& A, Frag& B) {
    if constexpr (C_IN >= 32) {
      constexpr int CC = C_IN / 32;
      const int tap = q / CC, cc = q % CC;
      const int dz = tap / 9, r9 = tap % 9, dy = r9 / 3, dx = r9 % 3;
      const unsigned short* ap = tapAddr(dz, dy, dx, (hi << 3) + cc * 32);
      A.u4[0] = *(const uint4*)(ap);
      A.u4[1] = *(const uint4*)(ap + 16);
    } else if constexpr (C_IN == 16) {
      const int tap0 = 2 * q, tap1 = 2 * q + 1;
      const unsigned short* ap0 =
          tapAddr(tap0 / 9, (tap0 % 9) / 3, tap0 % 3, hi << 3);
      const unsigned short* ap1 = (tap1 < 27)
          ? tapAddr(tap1 / 9, (tap1 % 9) / 3, tap1 % 3, hi << 3)
          : (in - ZPAD_EL);
      A.u4[0] = *(const uint4*)(ap0);
      A.u4[1] = *(const uint4*)(ap1);
    } else {  // C_IN == 4: per-pair gather
#pragma unroll
      for (int p = 0; p < 8; ++p) {
        int i   = p << 1;
        int ka  = (i & 7) + (hi << 3) + ((i >> 3) << 4);
        int k   = (q << 5) + ka;
        int tap = k >> CLOG;
        int c   = k & (C_IN - 1);
        int dz = tap / 9, r9 = tap % 9, dy = r9 / 3, dx = r9 % 3;
        const unsigned short* ap =
            (tap < 27) ? tapAddr(dz, dy, dx, c) : (in - ZPAD_EL);
        A.u[p] = *(const unsigned int*)ap;
      }
    }
    const unsigned int* bq = bp + (size_t)q * 256;
    B.u4[0] = *(const uint4*)(bq);
    B.u4[1] = *(const uint4*)(bq + 4);
  };

  // ---- double-buffered K loop: issue chunk q+1 loads before chunk q WMMA ----
  Frag A0, B0, A1, B1;
  loadAB(0, A0, B0);
#pragma unroll
  for (int q = 0; q < KCH; ++q) {
    Frag& Ac = (q & 1) ? A1 : A0;
    Frag& Bc = (q & 1) ? B1 : B0;
    Frag& An = (q & 1) ? A0 : A1;
    Frag& Bn = (q & 1) ? B0 : B1;
    if (q + 1 < KCH) loadAB(q + 1, An, Bn);
    acc = __builtin_amdgcn_wmma_f32_16x16x32_bf16(false, Ac.v, false, Bc.v,
                                                  (short)0, acc, false, false);
  }

  // ---- fused epilogue: y = relu(acc*s + b) * mask ----
  int o = (wv << 4) + (lane & 15);
  float sc = scale[o], sh = shift[o];
#pragma unroll
  for (int r = 0; r < 8; ++r) {
    int mm = r + (hi << 3);              // ISA f32 C/D layout: M = r (+8 for hi half)
    int x = x0 + mm;
    if (x < Wout) {
      size_t pidx = ((size_t)z * Hout + y) * Wout + x;
      float v = acc[r] * sc + sh;
      v = v > 0.f ? v : 0.f;
      v *= (float)mask[pidx];
      if (FINAL) {
        of32[(((size_t)o * Dout + z) * Hout + y) * (size_t)Wout + x] = v;
      } else {
        obf[pidx * C_OUT + o] = f2bf(v);
      }
    }
  }
}

// ---------------------------------------------------------------------------
// Host driver
// ---------------------------------------------------------------------------
static inline size_t conv_blocks(int D, int H, int W) {
  return (size_t)((W + 15) / 16) * (size_t)H * (size_t)D;
}

extern "C" void kernel_launch(void* const* d_in, const int* in_sizes, int n_in,
                              void* d_out, int out_size, void* d_ws, size_t ws_size,
                              hipStream_t stream) {
  (void)n_in; (void)out_size; (void)ws_size;

  const float* feats  = (const float*)d_in[0];
  const int*   coords = (const int*)d_in[1];
  // weights / scales / shifts in layer order
  const float* W[9] = {(const float*)d_in[3],  (const float*)d_in[6],  (const float*)d_in[9],
                       (const float*)d_in[12], (const float*)d_in[15], (const float*)d_in[18],
                       (const float*)d_in[21], (const float*)d_in[24], (const float*)d_in[27]};
  const float* SC[9] = {(const float*)d_in[4],  (const float*)d_in[7],  (const float*)d_in[10],
                        (const float*)d_in[13], (const float*)d_in[16], (const float*)d_in[19],
                        (const float*)d_in[22], (const float*)d_in[25], (const float*)d_in[28]};
  const float* SH[9] = {(const float*)d_in[5],  (const float*)d_in[8],  (const float*)d_in[11],
                        (const float*)d_in[14], (const float*)d_in[17], (const float*)d_in[20],
                        (const float*)d_in[23], (const float*)d_in[26], (const float*)d_in[29]};

  char* base = (char*)d_ws;
  size_t off = 0;
  auto alloc = [&](size_t bytes) -> char* {
    char* p = base + off;
    off = (off + bytes + 255) & ~(size_t)255;
    return p;
  };

  const size_t V0 = 41UL * 256 * 256;   // full res voxels
  const size_t V2 = 21UL * 128 * 128;
  const size_t V3 = 11UL * 64 * 64;
  const size_t V4 = 21UL * 127 * 127;

  // Pools: 256B zeroed prefix + tensor area.
  char* regA = alloc(256 + V0 * 32 * 2);   // dense, x2a, x3a, x4a, x5a
  char* regB = alloc(256 + V0 * 16 * 2);   // x1, x2b, x3b, x4b
  unsigned short* poolA = (unsigned short*)(regA + 256);
  unsigned short* poolB = (unsigned short*)(regB + 256);
  unsigned char* m0 = (unsigned char*)alloc(V0);
  unsigned char* m1 = (unsigned char*)alloc(V0);
  unsigned char* m2 = (unsigned char*)alloc(V2);
  unsigned char* m3 = (unsigned char*)alloc(V3);
  unsigned char* m4 = (unsigned char*)alloc(V4);
  unsigned char* m5 = (unsigned char*)alloc(V0);

  const int cins[9]   = {4, 16, 32, 32, 64, 64, 32, 32, 32};
  const int couts[9]  = {16, 32, 32, 64, 64, 32, 32, 32, 32};
  const int transp[9] = {0, 0, 0, 0, 0, 1, 0, 1, 0};
  unsigned short* frags[9];
  int kchs[9];
  for (int l = 0; l < 9; ++l) {
    kchs[l] = (27 * cins[l] + 31) / 32;
    frags[l] = (unsigned short*)alloc((size_t)(couts[l] / 16) * kchs[l] * 512 * 2);
  }

  // zero pool prefixes, dense input grid, and m0
  hipMemsetAsync(regA, 0, 256 + V0 * 4 * 2, stream);   // prefix + dense grid
  hipMemsetAsync(regB, 0, 256, stream);                // prefix only
  hipMemsetAsync(m0, 0, V0, stream);

  int nvox = in_sizes[0] / 4;
  scatter_kernel<<<(nvox + 255) / 256, 256, 0, stream>>>(
      feats, coords, poolA, m0, nvox);

  // mask pyramid
  {
    auto g = [](size_t n) { return (unsigned)((n + 255) / 256); };
    mask_dilate<0><<<g(V0), 256, 0, stream>>>(m0, m1, 41, 256, 256, 41, 256, 256);
    mask_dilate<1><<<g(V2), 256, 0, stream>>>(m1, m2, 41, 256, 256, 21, 128, 128);
    mask_dilate<1><<<g(V3), 256, 0, stream>>>(m2, m3, 21, 128, 128, 11, 64, 64);
    mask_dilate<2><<<g(V4), 256, 0, stream>>>(m3, m4, 11, 64, 64, 21, 127, 127);
    mask_dilate<2><<<g(V0), 256, 0, stream>>>(m4, m5, 21, 127, 127, 41, 256, 256);
  }

  // weight repacking into WMMA B-fragment layout
  for (int l = 0; l < 9; ++l) {
    int total = (couts[l] / 16) * kchs[l] * 512;
    repack_weights<<<(total + 255) / 256, 256, 0, stream>>>(
        W[l], frags[l], cins[l], couts[l], kchs[l], couts[l] / 16, transp[l]);
  }

  // ---- conv pipeline (pools alternate A<->B) ----
  // L1: dense(A,4ch) -> x1(B,16ch), stride1, full res
  conv3d_wmma<4, 16, 0, false><<<conv_blocks(41, 256, 256), 32, 0, stream>>>(
      poolA, (const unsigned int*)frags[0], SC[0], SH[0], m1, poolB, nullptr,
      41, 256, 256, 41, 256, 256);
  // L2a: x1(B,16ch) -> x2a(A,32ch), stride2
  conv3d_wmma<16, 32, 1, false><<<conv_blocks(21, 128, 128), 64, 0, stream>>>(
      poolB, (const unsigned int*)frags[1], SC[1], SH[1], m2, poolA, nullptr,
      41, 256, 256, 21, 128, 128);
  // L2b: x2a(A) -> x2b(B), stride1
  conv3d_wmma<32, 32, 0, false><<<conv_blocks(21, 128, 128), 64, 0, stream>>>(
      poolA, (const unsigned int*)frags[2], SC[2], SH[2], m2, poolB, nullptr,
      21, 128, 128, 21, 128, 128);
  // L3a: x2b(B) -> x3a(A,64ch), stride2
  conv3d_wmma<32, 64, 1, false><<<conv_blocks(11, 64, 64), 128, 0, stream>>>(
      poolB, (const unsigned int*)frags[3], SC[3], SH[3], m3, poolA, nullptr,
      21, 128, 128, 11, 64, 64);
  // L3b: x3a(A) -> x3b(B), stride1
  conv3d_wmma<64, 64, 0, false><<<conv_blocks(11, 64, 64), 128, 0, stream>>>(
      poolA, (const unsigned int*)frags[4], SC[4], SH[4], m3, poolB, nullptr,
      11, 64, 64, 11, 64, 64);
  // L4a: tconv x3b(B,64ch) -> x4a(A,32ch)
  conv3d_wmma<64, 32, 2, false><<<conv_blocks(21, 127, 127), 64, 0, stream>>>(
      poolB, (const unsigned int*)frags[5], SC[5], SH[5], m4, poolA, nullptr,
      11, 64, 64, 21, 127, 127);
  // L4b: x4a(A) -> x4b(B), stride1
  conv3d_wmma<32, 32, 0, false><<<conv_blocks(21, 127, 127), 64, 0, stream>>>(
      poolA, (const unsigned int*)frags[6], SC[6], SH[6], m4, poolB, nullptr,
      21, 127, 127, 21, 127, 127);
  // L5a: tconv x4b(B) -> x5a(A,32ch, full res)
  conv3d_wmma<32, 32, 2, false><<<conv_blocks(41, 256, 256), 64, 0, stream>>>(
      poolB, (const unsigned int*)frags[7], SC[7], SH[7], m5, poolA, nullptr,
      21, 127, 127, 41, 256, 256);
  // L5b: x5a(A) -> d_out (fp32 NCDHW), stride1, FINAL
  conv3d_wmma<32, 32, 0, true><<<conv_blocks(41, 256, 256), 64, 0, stream>>>(
      poolA, (const unsigned int*)frags[8], SC[8], SH[8], m5, nullptr, (float*)d_out,
      41, 256, 256, 41, 256, 256);
}
